// NoisyTopKRouter_29738353557793
// MI455X (gfx1250) — matrix operand order repaired and checked by reading, not compiled, not run on hardware
//
#include <hip/hip_runtime.h>
#include <hip/hip_bf16.h>

typedef __attribute__((ext_vector_type(16))) _Float16 v16h;
typedef __attribute__((ext_vector_type(8)))  float    v8f;
typedef __attribute__((ext_vector_type(4)))  float    v4f;

#define EMBD  1024
#define NEXP  64
#define NTOK  65536   // B*S = 8*8192
#define TOPK  8

// ---------------------------------------------------------------------------
// Kernel 0: convert w_router (rows 0..63) and w_noise (rows 64..127) fp32 ->
// combined f16 weight buffer wb[128][1024] in workspace.
// ---------------------------------------------------------------------------
__global__ __launch_bounds__(256) void cvt_weights_kernel(
    const float* __restrict__ wr, const float* __restrict__ wn,
    _Float16* __restrict__ wb) {
  const int i = blockIdx.x * 256 + threadIdx.x;       // 0 .. 131071
  const float v = (i < NEXP * EMBD) ? wr[i] : wn[i - NEXP * EMBD];
  wb[i] = (_Float16)v;
}

// ---------------------------------------------------------------------------
// Kernel 1: WMMA GEMM  [NTOK x 1024] * [1024 x 128]  -> logits(64) | noise(64)
// fused with noisy = u * softplus(noise) + logits + bias, written to d_out.
// 8 waves/WG, each wave: 16 tokens x 128 cols (8 C-tiles of v8f).
// Weights staged in LDS in 128-wide K chunks (padded stride, conflict-free).
// All 8 B fragments preloaded per k-step so the 16 ds_load_b128 go out as a
// clause and the 8 WMMAs issue back-to-back with partial dscnt waits.
// ---------------------------------------------------------------------------
__global__ __launch_bounds__(256) void router_gemm_kernel(
    const float* __restrict__ x, const float* __restrict__ u,
    const float* __restrict__ bias, const _Float16* __restrict__ wb,
    float* __restrict__ noisy) {
  // 128 rows x (128 halfs + 8 pad) = 272 bytes/row -> banks 4r..4r+3 per lane
  __shared__ __align__(16) unsigned char ldsw[128 * 272];

  const int tid  = threadIdx.x;
  const int lane = tid & 31;
  const int wave = tid >> 5;
  const int hs   = lane >> 4;    // half-wave select
  const int l15  = lane & 15;

  const int m0 = (blockIdx.x * 8 + wave) * 16;        // token tile base
  const float* xrow = x + (size_t)(m0 + l15) * EMBD;  // this lane's A row

  v8f acc[8];
  #pragma unroll
  for (int t = 0; t < 8; ++t)
    acc[t] = (v8f){0.f, 0.f, 0.f, 0.f, 0.f, 0.f, 0.f, 0.f};

  for (int kb = 0; kb < EMBD; kb += 128) {
    __syncthreads();
    // cooperative stage: wb[0..127][kb..kb+127] -> LDS (coalesced b128)
    #pragma unroll
    for (int ii = 0; ii < 8; ++ii) {
      const int i = ii * 256 + tid;                   // 0..2047
      const int r = i >> 4, c = i & 15;
      *(v4f*)(ldsw + r * 272 + c * 16) =
          *(const v4f*)(wb + (size_t)r * EMBD + kb + c * 8);
    }
    __syncthreads();

    #pragma unroll
    for (int kk = 0; kk < 128; kk += 32) {
      // ---- A fragment: 16-bit A 16x32 lane layout ----
      // elems 0..7  : K = k0 + hs*8 + j
      // elems 8..15 : K = k0 + 16 + hs*8 + (j-8)
      const v4f a0 = __builtin_nontemporal_load(
          (const v4f*)(xrow + kb + kk + hs * 8));
      const v4f a1 = __builtin_nontemporal_load(
          (const v4f*)(xrow + kb + kk + hs * 8 + 4));
      const v4f a2 = __builtin_nontemporal_load(
          (const v4f*)(xrow + kb + kk + 16 + hs * 8));
      const v4f a3 = __builtin_nontemporal_load(
          (const v4f*)(xrow + kb + kk + 16 + hs * 8 + 4));
      v16h A;
      A[0]  = (_Float16)a0.x; A[1]  = (_Float16)a0.y;
      A[2]  = (_Float16)a0.z; A[3]  = (_Float16)a0.w;
      A[4]  = (_Float16)a1.x; A[5]  = (_Float16)a1.y;
      A[6]  = (_Float16)a1.z; A[7]  = (_Float16)a1.w;
      A[8]  = (_Float16)a2.x; A[9]  = (_Float16)a2.y;
      A[10] = (_Float16)a2.z; A[11] = (_Float16)a2.w;
      A[12] = (_Float16)a3.x; A[13] = (_Float16)a3.y;
      A[14] = (_Float16)a3.z; A[15] = (_Float16)a3.w;

      // ---- preload ALL 8 B fragments (16x ds_load_b128, one clause) ----
      // B lane layout: N = l15, elems 0..7: K = hs*8 + j, elems 8..15: K = 16 + hs*8 + (j-8)
      v4f bf[16];
      #pragma unroll
      for (int t = 0; t < 8; ++t) {
        const unsigned char* bp =
            ldsw + (t * 16 + l15) * 272 + (kk + hs * 8) * 2;
        bf[2 * t + 0] = *(const v4f*)(bp);
        bf[2 * t + 1] = *(const v4f*)(bp + 32);
      }

      // ---- 8 back-to-back WMMAs ----
      #pragma unroll
      for (int t = 0; t < 8; ++t) {
        union { v16h h; v4f f[2]; } B;
        B.f[0] = bf[2 * t + 0];
        B.f[1] = bf[2 * t + 1];
        acc[t] = __builtin_amdgcn_wmma_f32_16x16x32_f16(
            false, A, false, B.h, (short)0, acc[t], false, false);
      }
    }
  }

  // ---- fused epilogue: noisy = u*softplus(noise) + logit + bias ----
  // C layout: tile t, VGPR r, lane: token = m0 + r + 8*hs, expert = 16t + l15
  #pragma unroll
  for (int t = 0; t < 4; ++t) {
    const int e = t * 16 + l15;
    const float be = bias[e];
    #pragma unroll
    for (int r = 0; r < 8; ++r) {
      const int tok = m0 + r + 8 * hs;
      const float logit = acc[t][r];
      const float nz    = acc[t + 4][r];
      const float uu    = u[(size_t)tok * NEXP + e];
      // numerically stable softplus
      const float sp = fmaxf(nz, 0.f) + __logf(1.f + __expf(-fabsf(nz)));
      noisy[(size_t)tok * NEXP + e] = fmaf(uu, sp, logit + be);
    }
  }
}

// ---------------------------------------------------------------------------
// Kernel 2: per-token top-8 of 64, softmax over top-8, scatter to prob row
// (in place over the noisy buffer) + int32 indices.
// ---------------------------------------------------------------------------
__global__ __launch_bounds__(256) void topk_kernel(
    float* __restrict__ probs, int* __restrict__ indices) {
  const int tok = blockIdx.x * 256 + threadIdx.x;
  float* row = probs + (size_t)tok * NEXP;

  v4f c[16];
  #pragma unroll
  for (int i = 0; i < 16; ++i) c[i] = ((const v4f*)row)[i];

  float arr[64];
  #pragma unroll
  for (int i = 0; i < 16; ++i) {
    arr[4 * i + 0] = c[i].x; arr[4 * i + 1] = c[i].y;
    arr[4 * i + 2] = c[i].z; arr[4 * i + 3] = c[i].w;
  }

  // descending 8-deep insertion (strict > : earlier index wins ties, as top_k)
  float val[TOPK]; int idx[TOPK];
  #pragma unroll
  for (int s = 0; s < TOPK; ++s) { val[s] = -__builtin_inff(); idx[s] = 0; }
  #pragma unroll
  for (int j = 0; j < 64; ++j) {
    float cv = arr[j]; int ci = j;
    #pragma unroll
    for (int s = 0; s < TOPK; ++s) {
      const bool gt = cv > val[s];
      const float tv = val[s]; const int ti = idx[s];
      val[s] = gt ? cv : tv;
      idx[s] = gt ? ci : ti;
      cv = gt ? tv : cv;
      ci = gt ? ti : ci;
    }
  }

  // softmax over the top-8
  const float m = val[0];
  float p[TOPK], sum = 0.f;
  #pragma unroll
  for (int k = 0; k < TOPK; ++k) { p[k] = __expf(val[k] - m); sum += p[k]; }
  const float inv = 1.f / sum;

  // zero row, then scatter the 8 probabilities (same-thread ordering)
  const v4f z = (v4f){0.f, 0.f, 0.f, 0.f};
  #pragma unroll
  for (int i = 0; i < 16; ++i) ((v4f*)row)[i] = z;
  #pragma unroll
  for (int k = 0; k < TOPK; ++k) row[idx[k]] = p[k] * inv;
  #pragma unroll
  for (int k = 0; k < TOPK; ++k) indices[(size_t)tok * TOPK + k] = idx[k];
}

// ---------------------------------------------------------------------------
extern "C" void kernel_launch(void* const* d_in, const int* in_sizes, int n_in,
                              void* d_out, int out_size, void* d_ws,
                              size_t ws_size, hipStream_t stream) {
  const float* x    = (const float*)d_in[0];  // [8,8192,1024]
  const float* u    = (const float*)d_in[1];  // [8,8192,64]
  const float* wr   = (const float*)d_in[2];  // [64,1024]
  const float* wn   = (const float*)d_in[3];  // [64,1024]
  const float* bias = (const float*)d_in[4];  // [64]
  (void)in_sizes; (void)n_in; (void)out_size; (void)ws_size;

  _Float16* wb = (_Float16*)d_ws;                       // 128*1024 f16 = 256 KB
  float* noisy_then_probs = (float*)d_out;              // NTOK*64 f32
  int* indices = (int*)d_out + (size_t)NTOK * NEXP;     // NTOK*8 i32

  cvt_weights_kernel<<<(128 * EMBD) / 256, 256, 0, stream>>>(wr, wn, wb);
  router_gemm_kernel<<<NTOK / 128, 256, 0, stream>>>(x, u, bias, wb,
                                                     noisy_then_probs);
  topk_kernel<<<NTOK / 256, 256, 0, stream>>>(noisy_then_probs, indices);
}
